// CharRNN_70317204570246
// MI455X (gfx1250) — matrix operand order repaired and checked by reading, not compiled, not run on hardware
//
#include <hip/hip_runtime.h>
#include <hip/hip_bf16.h>

#define VOCAB  4096
#define HIDDEN 512
#define BATCH  64
#define TLEN   512

#define HSTRIDE 520   // h rows padded by 16B: bank step 4 -> conflict-minimal
#define KSTRIDE 40    // FC LDS tile rows (32 K + 8 pad): bank step 20

typedef __attribute__((ext_vector_type(16))) __bf16 v16bf;
typedef __attribute__((ext_vector_type(8)))  float  v8f;

union Frag { v16bf v; uint4 q[2]; };
union CFrag { v8f v; uint4 q[2]; };

// native V_TANH_F32 if the toolchain exposes it; else OCML fallback
#if __has_builtin(__builtin_amdgcn_tanhf)
  #define FAST_TANH(x) __builtin_amdgcn_tanhf(x)
#elif __has_builtin(__builtin_amdgcn_tanh_f32)
  #define FAST_TANH(x) __builtin_amdgcn_tanh_f32(x)
#else
  #define FAST_TANH(x) tanhf(x)
#endif

// ---------------------------------------------------------------------------
// Prep: transpose W_ih [H][V] -> W_ihT [V][H] (f32), and convert
// W_hh [H][H], W_fc [V][H] to bf16 row-major (K-contiguous for WMMA B frags).
// ---------------------------------------------------------------------------
__global__ __launch_bounds__(256) void prep_kernel(
    const float* __restrict__ W_ih, const float* __restrict__ W_hh,
    const float* __restrict__ W_fc,
    float* __restrict__ W_ihT, __bf16* __restrict__ whh_bf,
    __bf16* __restrict__ wfc_bf)
{
    unsigned i = blockIdx.x * 256u + threadIdx.x;
    const unsigned NT = VOCAB * HIDDEN;
    const unsigned NH = HIDDEN * HIDDEN;
    if (i < NT) {
        unsigned h = i / VOCAB, v = i % VOCAB;   // coalesced read of W_ih
        W_ihT[v * HIDDEN + h] = W_ih[i];
    } else if (i < NT + NH) {
        unsigned j = i - NT;
        whh_bf[j] = (__bf16)W_hh[j];
    } else if (i < NT + NH + NT) {
        unsigned j = i - NT - NH;
        wfc_bf[j] = (__bf16)W_fc[j];
    }
}

// ---------------------------------------------------------------------------
// Embedding gather, written directly in WMMA C-fragment order:
//   xsw[((t*128 + mt*32 + c)*32 + lane)*8 + r]
//     = value for row (b = mt*16 + (lane>>4)*8 + r), col (h = c*16 + lane&15)
// so the RNN seeds each accumulator with two b128 loads.
// One thread produces 8 r-values (32B contiguous store); W_ihT gathers hit L2.
// ---------------------------------------------------------------------------
__global__ __launch_bounds__(256) void embed_kernel(
    const int* __restrict__ tok, const float* __restrict__ W_ihT,
    const float* __restrict__ b_ih, const float* __restrict__ b_hh,
    float* __restrict__ xsw)
{
    __shared__ int toks[BATCH];
    unsigned j    = blockIdx.x * 256u + threadIdx.x;
    unsigned lane = j & 31u;
    unsigned c    = (j >> 5) & 31u;    // column tile 0..31
    unsigned mt   = (j >> 10) & 3u;    // row tile 0..3
    unsigned t    = j >> 12;           // timestep (constant per block)

    if (threadIdx.x < BATCH)
        toks[threadIdx.x] = tok[(size_t)threadIdx.x * TLEN + t];
    __syncthreads();

    unsigned col  = c * 16u + (lane & 15u);
    unsigned half = lane >> 4;
    float bias = b_ih[col] + b_hh[col];

    float vals[8];
    #pragma unroll
    for (int r = 0; r < 8; ++r) {
        unsigned row = mt * 16u + half * 8u + (unsigned)r;   // batch index
        unsigned tk  = (unsigned)toks[row];
        vals[r] = W_ihT[(size_t)tk * HIDDEN + col] + bias;
    }
    float* dst = xsw + (size_t)((t * 128u + mt * 32u + c) * 32u + lane) * 8u;
    *(float4*)(dst)     = make_float4(vals[0], vals[1], vals[2], vals[3]);
    *(float4*)(dst + 4) = make_float4(vals[4], vals[5], vals[6], vals[7]);
}

// ---------------------------------------------------------------------------
// RNN scan: one 256-thread workgroup (8 waves). Wave w owns a 64-wide N
// stripe; register tile 64(M) x 64(N) = 4x4 WMMA tiles (128 acc VGPRs).
// h kept bf16 in LDS (padded rows). Seeds via swizzled b128 loads; native
// v_tanh; out_h written by coalesced LDS->global b128 copy after barrier.
// ---------------------------------------------------------------------------
__global__ __launch_bounds__(256) void rnn_kernel(
    const float* __restrict__ h0, const float* __restrict__ xsw,
    const __bf16* __restrict__ whh_bf, __bf16* __restrict__ outh_bf,
    float* __restrict__ hlast)
{
    __shared__ __bf16 hbuf[BATCH * HSTRIDE];   // 65 KB, padded rows

    const unsigned tid = threadIdx.x;
    for (unsigned i = tid; i < BATCH * HIDDEN; i += 256u)
        hbuf[(i >> 9) * HSTRIDE + (i & 511u)] = (__bf16)h0[i];
    __syncthreads();

    const unsigned lane  = tid & 31u;
    const unsigned w     = tid >> 5;                 // wave 0..7
    const unsigned n0    = w * 64u;                  // N stripe base
    const unsigned nn    = lane & 15u;
    const unsigned mm    = lane & 15u;               // A-frag row
    const unsigned mBase = (lane < 16u) ? 0u : 8u;   // C/D row half
    const unsigned kA    = (lane < 16u) ? 0u : 16u;  // A K half offset (elems)
    const unsigned kB    = (lane < 16u) ? 0u : 16u;  // B K half offset (elems)

    #pragma unroll 1
    for (int t = 0; t < TLEN; ++t) {
        const float* xt = xsw + (size_t)t * (BATCH * HIDDEN);

        // prefetch next step's x block (1024 cachelines of 128B)
        if (t + 1 < TLEN) {
            const float* xnext = xt + BATCH * HIDDEN;
            #pragma unroll
            for (int j = 0; j < 4; ++j)
                __builtin_prefetch(xnext + ((size_t)tid + j * 256u) * 32u, 0, 0);
        }

        // seed accumulators from swizzled x: 2 x b128 per tile
        v8f acc[4][4];
        #pragma unroll
        for (int mt = 0; mt < 4; ++mt)
            #pragma unroll
            for (int nt = 0; nt < 4; ++nt) {
                const uint4* p = (const uint4*)(
                    xt + (size_t)((mt * 32u + (w * 4u + nt)) * 32u + lane) * 8u);
                CFrag cf; cf.q[0] = p[0]; cf.q[1] = p[1];
                acc[mt][nt] = cf.v;
            }

        #pragma unroll 1
        for (int kt = 0; kt < 16; ++kt) {
            const unsigned k0 = kt * 32u;
            Frag b[4];   // B = W_hh^T tiles: B[k][n] = W_hh[n][k]
            #pragma unroll
            for (int nt = 0; nt < 4; ++nt) {
                const __bf16* bp =
                    whh_bf + (size_t)(n0 + nt * 16u + nn) * HIDDEN + k0 + kB;
                b[nt].q[0] = *(const uint4*)(bp);
                b[nt].q[1] = *(const uint4*)(bp + 8);
            }
            #pragma unroll
            for (int mt = 0; mt < 4; ++mt) {
                Frag a;   // A = h tile from padded LDS
                const __bf16* ap = &hbuf[(mt * 16u + mm) * HSTRIDE + k0];
                a.q[0] = *(const uint4*)(ap + kA);
                a.q[1] = *(const uint4*)(ap + 8 + kA);
                #pragma unroll
                for (int nt = 0; nt < 4; ++nt)
                    acc[mt][nt] = __builtin_amdgcn_wmma_f32_16x16x32_bf16(
                        false, a.v, false, b[nt].v, (short)0, acc[mt][nt],
                        false, false);
            }
        }

        __syncthreads();   // all LDS reads of h_{t-1} complete

        // h_t = tanh(acc) -> LDS (single v_tanh_f32 per element)
        #pragma unroll
        for (int mt = 0; mt < 4; ++mt)
            #pragma unroll
            for (int nt = 0; nt < 4; ++nt)
                #pragma unroll
                for (int r = 0; r < 8; ++r) {
                    unsigned row = mt * 16u + mBase + (unsigned)r;
                    unsigned col = n0 + nt * 16u + nn;
                    hbuf[row * HSTRIDE + col] = (__bf16)FAST_TANH(acc[mt][nt][r]);
                }
        __syncthreads();   // h_t complete in LDS

        // coalesced copy-out: h_t -> outh[t] (16 x b128 per thread)
        {
            __bf16* dstT = outh_bf + (size_t)t * (BATCH * HIDDEN);
            #pragma unroll
            for (int k2 = 0; k2 < 16; ++k2) {
                unsigned s   = tid + (unsigned)k2 * 256u;   // 8-elem segments
                unsigned row = s >> 6;
                unsigned cc  = (s & 63u) * 8u;
                *(uint4*)(dstT + row * HIDDEN + cc) =
                    *(const uint4*)&hbuf[row * HSTRIDE + cc];
            }
        }
        // next iteration's K-loop only *reads* hbuf; the post-K barrier
        // orders these copy reads against the next write phase.
    }

    __syncthreads();
    for (unsigned i = tid; i < BATCH * HIDDEN; i += 256u)
        hlast[i] = (float)hbuf[(i >> 9) * HSTRIDE + (i & 511u)];
}

// ---------------------------------------------------------------------------
// FC: logits[b][t][v] = out_h[(t,b)] . W_fc[v] + b_fc[v].
// 512-thread WGs compute 256x256 tiles; K staged through LDS in 32-steps
// (coalesced 32B/lane loads, padded rows). 16 waves, each 64x64 register
// tile = 16 WMMA accumulators, K-loop unroll 1.
// ---------------------------------------------------------------------------
__global__ __launch_bounds__(512) void fc_kernel(
    const __bf16* __restrict__ A, const __bf16* __restrict__ Wfc,
    const float* __restrict__ b_fc, float* __restrict__ out)
{
    __shared__ __bf16 As[256 * KSTRIDE];   // 20 KB
    __shared__ __bf16 Bs[256 * KSTRIDE];   // 20 KB

    const unsigned tid   = threadIdx.x;
    const unsigned lane  = tid & 31u;
    const unsigned w     = tid >> 5;          // 0..15
    const unsigned mw    = w & 3u;            // M sub-block (x64)
    const unsigned nw    = w >> 2;            // N sub-block (x64)
    const unsigned m0    = blockIdx.y * 256u;
    const unsigned n0    = blockIdx.x * 256u;
    const unsigned nn    = lane & 15u;
    const unsigned mm    = lane & 15u;
    const unsigned mBase = (lane < 16u) ? 0u : 8u;
    const unsigned kA    = (lane < 16u) ? 0u : 16u;
    const unsigned kB    = (lane < 16u) ? 0u : 16u;

    const unsigned lr = tid >> 1;            // staging row 0..255
    const unsigned kh = (tid & 1u) * 16u;    // staging K half (elems)

    v8f acc[4][4] = {};

    #pragma unroll 1
    for (int kt = 0; kt < 16; ++kt) {
        const unsigned k0 = kt * 32u;

        // stage A and B K-slabs: one 32B coalesced load per lane each
        {
            const uint4* ga = (const uint4*)(A   + (size_t)(m0 + lr) * HIDDEN + k0 + kh);
            const uint4* gb = (const uint4*)(Wfc + (size_t)(n0 + lr) * HIDDEN + k0 + kh);
            uint4 va0 = ga[0], va1 = ga[1];
            uint4 vb0 = gb[0], vb1 = gb[1];
            uint4* sa = (uint4*)&As[lr * KSTRIDE + kh];
            uint4* sb = (uint4*)&Bs[lr * KSTRIDE + kh];
            sa[0] = va0; sa[1] = va1;
            sb[0] = vb0; sb[1] = vb1;
        }
        __syncthreads();

        Frag b[4];
        #pragma unroll
        for (int nt = 0; nt < 4; ++nt) {
            const __bf16* bp = &Bs[(nw * 64u + nt * 16u + nn) * KSTRIDE + kB];
            b[nt].q[0] = *(const uint4*)(bp);
            b[nt].q[1] = *(const uint4*)(bp + 8);
        }
        #pragma unroll
        for (int mt = 0; mt < 4; ++mt) {
            Frag a;
            const __bf16* ap = &As[(mw * 64u + mt * 16u + mm) * KSTRIDE + kA];
            a.q[0] = *(const uint4*)(ap);
            a.q[1] = *(const uint4*)(ap + 8);
            #pragma unroll
            for (int nt = 0; nt < 4; ++nt)
                acc[mt][nt] = __builtin_amdgcn_wmma_f32_16x16x32_bf16(
                    false, a.v, false, b[nt].v, (short)0, acc[mt][nt],
                    false, false);
        }
        __syncthreads();   // LDS consumed; safe to overwrite next iter
    }

    #pragma unroll
    for (int nt = 0; nt < 4; ++nt) {
        unsigned col = n0 + nw * 64u + nt * 16u + nn;
        float bias = b_fc[col];
        #pragma unroll
        for (int mt = 0; mt < 4; ++mt) {
            #pragma unroll
            for (int r = 0; r < 8; ++r) {
                unsigned gm = m0 + mw * 64u + mt * 16u + mBase + (unsigned)r;
                unsigned tt = gm >> 6, bb = gm & 63u;
                out[(size_t)bb * ((size_t)TLEN * VOCAB) + (size_t)tt * VOCAB + col]
                    = acc[mt][nt][r] + bias;
            }
        }
    }
}

// ---------------------------------------------------------------------------
extern "C" void kernel_launch(void* const* d_in, const int* in_sizes, int n_in,
                              void* d_out, int out_size, void* d_ws, size_t ws_size,
                              hipStream_t stream)
{
    (void)in_sizes; (void)n_in; (void)out_size; (void)ws_size;

    const int*   input  = (const int*)  d_in[0];
    const float* hidden = (const float*)d_in[1];
    const float* W_ih   = (const float*)d_in[2];
    const float* W_hh   = (const float*)d_in[3];
    const float* b_ih   = (const float*)d_in[4];
    const float* b_hh   = (const float*)d_in[5];
    const float* W_fc   = (const float*)d_in[6];
    const float* b_fc   = (const float*)d_in[7];

    // workspace layout (bytes): ~108.5 MB total
    char* ws = (char*)d_ws;
    const size_t X_BYTES    = (size_t)TLEN * BATCH * HIDDEN * 4;   // 64 MB
    const size_t OUTH_BYTES = (size_t)TLEN * BATCH * HIDDEN * 2;   // 32 MB
    const size_t WIHT_BYTES = (size_t)VOCAB * HIDDEN * 4;          //  8 MB
    const size_t WHH_BYTES  = (size_t)HIDDEN * HIDDEN * 2;         // .5 MB

    float*  xsw    = (float*) (ws);
    __bf16* outh   = (__bf16*)(ws + X_BYTES);
    float*  W_ihT  = (float*) (ws + X_BYTES + OUTH_BYTES);
    __bf16* whh_bf = (__bf16*)(ws + X_BYTES + OUTH_BYTES + WIHT_BYTES);
    __bf16* wfc_bf = (__bf16*)(ws + X_BYTES + OUTH_BYTES + WIHT_BYTES + WHH_BYTES);

    float* out   = (float*)d_out;
    float* hlast = out + (size_t)BATCH * TLEN * VOCAB;

    {
        unsigned total = VOCAB * HIDDEN + HIDDEN * HIDDEN + VOCAB * HIDDEN;
        prep_kernel<<<(total + 255u) / 256u, 256, 0, stream>>>(
            W_ih, W_hh, W_fc, W_ihT, whh_bf, wfc_bf);
    }
    {
        unsigned total = (TLEN * BATCH * HIDDEN) / 8;   // 2,097,152 threads
        embed_kernel<<<total / 256u, 256, 0, stream>>>(
            input, W_ihT, b_ih, b_hh, xsw);
    }
    rnn_kernel<<<1, 256, 0, stream>>>(hidden, xsw, whh_bf, outh, hlast);
    {
        dim3 grid(VOCAB / 256, (TLEN * BATCH) / 256);   // (16, 128)
        fc_kernel<<<grid, 512, 0, stream>>>(outh, wfc_bf, b_fc, out);
    }
}
